// Mamba3DBlock_70153995813183
// MI455X (gfx1250) — matrix operand order repaired and checked by reading, not compiled, not run on hardware
//
#include <hip/hip_runtime.h>
#include <hip/hip_bf16.h>

#define BATCH   2
#define LSEQ    2048
#define DMODEL  1024
#define DINNER  1536
#define DSTATE  8
#define DTRANK  64
#define DFF     2048
#define NTOK    (BATCH * LSEQ)          // 4096
#define XDBL_W  (DTRANK + 2 * DSTATE)   // 80

typedef __attribute__((ext_vector_type(16))) _Float16 v16h;
typedef __attribute__((ext_vector_type(8)))  _Float16 v8h;
typedef __attribute__((ext_vector_type(8)))  float    v8f;

// ---------------------------------------------------------------------------
// fp32 -> fp16 conversion
// ---------------------------------------------------------------------------
__global__ void cvt_f32_f16_kernel(const float* __restrict__ in,
                                   _Float16* __restrict__ out, int n) {
    int i = blockIdx.x * blockDim.x + threadIdx.x;
    if (i < n) out[i] = (_Float16)in[i];
}

// ---------------------------------------------------------------------------
// LayerNorm over last dim D (one block per row), writes f16 for GEMM input
// ---------------------------------------------------------------------------
__global__ __launch_bounds__(256) void ln_kernel(const float* __restrict__ x,
                                                 const float* __restrict__ g,
                                                 const float* __restrict__ b,
                                                 _Float16* __restrict__ out,
                                                 int D) {
    int row = blockIdx.x;
    const float* xr = x + (size_t)row * D;
    float s = 0.f, s2 = 0.f;
    for (int i = threadIdx.x; i < D; i += blockDim.x) {
        float v = xr[i];
        s += v; s2 += v * v;
    }
    __shared__ float r0[256];
    __shared__ float r1[256];
    int tid = threadIdx.x;
    r0[tid] = s; r1[tid] = s2;
    __syncthreads();
    for (int off = 128; off > 0; off >>= 1) {
        if (tid < off) { r0[tid] += r0[tid + off]; r1[tid] += r1[tid + off]; }
        __syncthreads();
    }
    float mu  = r0[0] / (float)D;
    float var = r1[0] / (float)D - mu * mu;
    float inv = rsqrtf(var + 1e-5f);
    for (int i = threadIdx.x; i < D; i += blockDim.x) {
        float v = (xr[i] - mu) * inv * g[i] + b[i];
        out[(size_t)row * D + i] = (_Float16)v;
    }
}

// ---------------------------------------------------------------------------
// WMMA fragment loads (this code's contiguous k order, k = 16*(lane/16) + s, same for A and B; wave32)
//   A 16x32: lanes 0-15 = rows M, lane-group selects K {0..15}/{16..31}
//   B 32x16: lanes 0-15 = cols N, lane-group selects K {0..15}/{16..31}; since
//            we compute X*W^T, a B fragment is 16 contiguous K-values of row n of W.
// ---------------------------------------------------------------------------
__device__ __forceinline__ v16h load_fragA(const _Float16* __restrict__ p, int ld,
                                           int rowBase, int kk, int lane) {
    int r  = rowBase + (lane & 15);
    int kb = kk + ((lane >> 4) << 4);
    const _Float16* base = p + (size_t)r * ld + kb;
    v8h lo = *(const v8h*)(base);
    v8h hi = *(const v8h*)(base + 8);
    v16h f;
#pragma unroll
    for (int i = 0; i < 8; ++i) { f[i] = lo[i]; f[i + 8] = hi[i]; }
    return f;
}

__device__ __forceinline__ v16h load_fragB(const _Float16* __restrict__ p, int ld,
                                           int rowBase, int kk, int lane, int nMax) {
    int r = rowBase + (lane & 15);
    if (r > nMax) r = nMax;  // clamp for ragged N (reads stay in-bounds)
    int kb = kk + ((lane >> 4) << 4);
    return *(const v16h*)(p + (size_t)r * ld + kb);
}

// ---------------------------------------------------------------------------
// Generic WMMA GEMM: C[M,N] = act(A[M,K] * W[N,K]^T + bias) (+ residual)
// Block = 256 threads = 8 waves; block tile 128x128; wave tile 32x64
// (2 A-frags x 4 B-frags -> 8 WMMAs per 6 b128-load pairs per K-step).
// act: 0 = none, 1 = exact GELU, 2 = softplus
// ---------------------------------------------------------------------------
__global__ __launch_bounds__(256) void wmma_gemm_kernel(
    const _Float16* __restrict__ A, int lda,
    const _Float16* __restrict__ W, int ldw,
    const float* __restrict__ bias,
    const float* __restrict__ residual,
    float* __restrict__ Cf, _Float16* __restrict__ Ch, int ldc,
    int M, int N, int K, int act) {
    int lane  = threadIdx.x & 31;
    int wave  = threadIdx.x >> 5;
    int waveM = wave & 3;   // 4 waves over M
    int waveN = wave >> 2;  // 2 waves over N
    int mbase = blockIdx.y * 128 + waveM * 32;
    int nbase = blockIdx.x * 128 + waveN * 64;
    if (mbase >= M || nbase >= N) return;

    v8f acc[2][4];
#pragma unroll
    for (int i = 0; i < 2; ++i)
#pragma unroll
        for (int j = 0; j < 4; ++j) acc[i][j] = (v8f){};

    for (int kk = 0; kk < K; kk += 32) {
        v16h a0 = load_fragA(A, lda, mbase,      kk, lane);
        v16h a1 = load_fragA(A, lda, mbase + 16, kk, lane);
        v16h b0 = load_fragB(W, ldw, nbase,      kk, lane, N - 1);
        v16h b1 = load_fragB(W, ldw, nbase + 16, kk, lane, N - 1);
        v16h b2 = load_fragB(W, ldw, nbase + 32, kk, lane, N - 1);
        v16h b3 = load_fragB(W, ldw, nbase + 48, kk, lane, N - 1);
        acc[0][0] = __builtin_amdgcn_wmma_f32_16x16x32_f16(false, a0, false, b0,
                                                           (short)0, acc[0][0], false, false);
        acc[0][1] = __builtin_amdgcn_wmma_f32_16x16x32_f16(false, a0, false, b1,
                                                           (short)0, acc[0][1], false, false);
        acc[0][2] = __builtin_amdgcn_wmma_f32_16x16x32_f16(false, a0, false, b2,
                                                           (short)0, acc[0][2], false, false);
        acc[0][3] = __builtin_amdgcn_wmma_f32_16x16x32_f16(false, a0, false, b3,
                                                           (short)0, acc[0][3], false, false);
        acc[1][0] = __builtin_amdgcn_wmma_f32_16x16x32_f16(false, a1, false, b0,
                                                           (short)0, acc[1][0], false, false);
        acc[1][1] = __builtin_amdgcn_wmma_f32_16x16x32_f16(false, a1, false, b1,
                                                           (short)0, acc[1][1], false, false);
        acc[1][2] = __builtin_amdgcn_wmma_f32_16x16x32_f16(false, a1, false, b2,
                                                           (short)0, acc[1][2], false, false);
        acc[1][3] = __builtin_amdgcn_wmma_f32_16x16x32_f16(false, a1, false, b3,
                                                           (short)0, acc[1][3], false, false);
    }

    // C/D layout: lane&15 = N, lane>>4 selects M offset 0/8, vgpr v = M row v
    int nn = lane & 15;
    int mo = (lane >> 4) << 3;
#pragma unroll
    for (int ti = 0; ti < 2; ++ti) {
#pragma unroll
        for (int tj = 0; tj < 4; ++tj) {
            v8f d = acc[ti][tj];
            int n = nbase + tj * 16 + nn;
            if (n >= N) continue;
            float bv = bias ? bias[n] : 0.f;
#pragma unroll
            for (int v = 0; v < 8; ++v) {
                int m = mbase + ti * 16 + mo + v;
                float val = d[v] + bv;
                if (act == 1) {
                    val = 0.5f * val * (1.f + erff(val * 0.70710678118654752f));
                } else if (act == 2) {
                    val = (val > 20.f) ? val : log1pf(__expf(val));
                }
                if (residual) val += residual[(size_t)m * ldc + n];
                if (Cf) Cf[(size_t)m * ldc + n] = val;
                if (Ch) Ch[(size_t)m * ldc + n] = (_Float16)val;
            }
        }
    }
}

// ---------------------------------------------------------------------------
// Causal depthwise conv (width 4) + bias + SiLU; input = xi half of xz
// ---------------------------------------------------------------------------
__global__ void conv_silu_kernel(const float* __restrict__ xz,
                                 const float* __restrict__ cw,
                                 const float* __restrict__ cb,
                                 float* __restrict__ u32,
                                 _Float16* __restrict__ u16) {
    int idx = blockIdx.x * blockDim.x + threadIdx.x;
    if (idx >= NTOK * DINNER) return;
    int d = idx % DINNER;
    int t = idx / DINNER;
    int l = t % LSEQ;
    float acc = cb[d];
#pragma unroll
    for (int j = 0; j < 4; ++j) {
        int lp = l - 3 + j;
        if (lp >= 0)
            acc += cw[d * 4 + j] * xz[(size_t)(t - 3 + j) * (2 * DINNER) + d];
    }
    float s = acc / (1.f + __expf(-acc));  // SiLU
    u32[idx] = s;
    u16[idx] = (_Float16)s;
}

// ---------------------------------------------------------------------------
// Selective scan: one thread per (b, d) channel; 8 states in registers;
// software-pipelined loads (next step fetched during current compute) +
// prefetch a few steps ahead; fused +u*D, SiLU(z) gating, f16 output.
// ---------------------------------------------------------------------------
__global__ void scan_kernel(const float* __restrict__ delta,
                            const float* __restrict__ u,
                            const float* __restrict__ xdbl,
                            const float* __restrict__ xz,
                            const float* __restrict__ A_log,
                            const float* __restrict__ Dp,
                            _Float16* __restrict__ y16) {
    int idx = blockIdx.x * blockDim.x + threadIdx.x;
    if (idx >= BATCH * DINNER) return;
    int b = idx / DINNER;
    int d = idx % DINNER;
    float Arow[DSTATE], h[DSTATE];
#pragma unroll
    for (int n = 0; n < DSTATE; ++n) {
        Arow[n] = -__expf(A_log[d * DSTATE + n]);
        h[n] = 0.f;
    }
    float Dd = Dp[d];

    size_t t0 = (size_t)b * LSEQ;
    // pipeline stage: current values
    float dt_c = delta[t0 * DINNER + d];
    float uu_c = u[t0 * DINNER + d];
    float zz_c = xz[t0 * (2 * DINNER) + DINNER + d];
    float Bc[DSTATE], Cc[DSTATE];
#pragma unroll
    for (int n = 0; n < DSTATE; ++n) {
        Bc[n] = xdbl[t0 * XDBL_W + DTRANK + n];
        Cc[n] = xdbl[t0 * XDBL_W + DTRANK + DSTATE + n];
    }

    for (int l = 0; l < LSEQ; ++l) {
        // issue next-step loads first (independent of current compute)
        int ln = (l + 1 < LSEQ) ? (l + 1) : l;
        size_t tn = t0 + ln;
        float dt_n = delta[tn * DINNER + d];
        float uu_n = u[tn * DINNER + d];
        float zz_n = xz[tn * (2 * DINNER) + DINNER + d];
        float Bn[DSTATE], Cn[DSTATE];
#pragma unroll
        for (int n = 0; n < DSTATE; ++n) {
            Bn[n] = xdbl[tn * XDBL_W + DTRANK + n];
            Cn[n] = xdbl[tn * XDBL_W + DTRANK + DSTATE + n];
        }
        if (l + 8 < LSEQ) {
            __builtin_prefetch(&delta[(t0 + l + 8) * DINNER + d], 0, 0);
            __builtin_prefetch(&u[(t0 + l + 8) * DINNER + d], 0, 0);
        }

        // compute current step
        size_t t = t0 + l;
        float acc = 0.f;
#pragma unroll
        for (int n = 0; n < DSTATE; ++n) {
            float dA = __expf(dt_c * Arow[n]);
            h[n] = dA * h[n] + dt_c * Bc[n] * uu_c;
            acc += h[n] * Cc[n];
        }
        float yv = acc + uu_c * Dd;
        yv *= zz_c / (1.f + __expf(-zz_c));  // y * silu(z)
        y16[t * DINNER + d] = (_Float16)yv;

        // rotate pipeline
        dt_c = dt_n; uu_c = uu_n; zz_c = zz_n;
#pragma unroll
        for (int n = 0; n < DSTATE; ++n) { Bc[n] = Bn[n]; Cc[n] = Cn[n]; }
    }
}

// ---------------------------------------------------------------------------
// Host-side orchestration
// ---------------------------------------------------------------------------
extern "C" void kernel_launch(void* const* d_in, const int* in_sizes, int n_in,
                              void* d_out, int out_size, void* d_ws, size_t ws_size,
                              hipStream_t stream) {
    const float* x        = (const float*)d_in[0];
    const float* ln1_g    = (const float*)d_in[1];
    const float* ln1_b    = (const float*)d_in[2];
    const float* ln2_g    = (const float*)d_in[3];
    const float* ln2_b    = (const float*)d_in[4];
    const float* in_proj  = (const float*)d_in[5];   // [3072,1024]
    const float* conv_w   = (const float*)d_in[6];   // [1536,1,4]
    const float* conv_b   = (const float*)d_in[7];
    const float* x_proj   = (const float*)d_in[8];   // [80,1536]
    const float* dt_proj  = (const float*)d_in[9];   // [1536,64]
    const float* dt_b     = (const float*)d_in[10];
    const float* A_log    = (const float*)d_in[11];  // [1536,8]
    const float* Dp       = (const float*)d_in[12];
    const float* out_proj = (const float*)d_in[13];  // [1024,1536]
    const float* ffn_w1   = (const float*)d_in[14];  // [2048,1024]
    const float* ffn_b1   = (const float*)d_in[15];
    const float* ffn_w2   = (const float*)d_in[16];  // [1024,2048]
    const float* ffn_b2   = (const float*)d_in[17];
    float* out = (float*)d_out;

    char* wsp = (char*)d_ws;
    size_t off = 0;
    auto alloc = [&](size_t bytes) -> char* {
        char* p = wsp + off;
        off = (off + bytes + 255) & ~(size_t)255;
        return p;
    };

    _Float16* xp16   = (_Float16*)alloc((size_t)NTOK * DMODEL * 2);
    _Float16* w_in16 = (_Float16*)alloc((size_t)2 * DINNER * DMODEL * 2);
    _Float16* w_xp16 = (_Float16*)alloc((size_t)XDBL_W * DINNER * 2);
    _Float16* w_dt16 = (_Float16*)alloc((size_t)DINNER * DTRANK * 2);
    _Float16* w_op16 = (_Float16*)alloc((size_t)DMODEL * DINNER * 2);
    _Float16* w_f116 = (_Float16*)alloc((size_t)DFF * DMODEL * 2);
    _Float16* w_f216 = (_Float16*)alloc((size_t)DMODEL * DFF * 2);
    float*    xz     = (float*)   alloc((size_t)NTOK * 2 * DINNER * 4);
    float*    u32    = (float*)   alloc((size_t)NTOK * DINNER * 4);
    _Float16* u16    = (_Float16*)alloc((size_t)NTOK * DINNER * 2);
    float*    xdbl   = (float*)   alloc((size_t)NTOK * XDBL_W * 4);
    _Float16* xdbl16 = (_Float16*)alloc((size_t)NTOK * XDBL_W * 2);
    float*    delta  = (float*)   alloc((size_t)NTOK * DINNER * 4);
    _Float16* y16    = (_Float16*)alloc((size_t)NTOK * DINNER * 2);
    float*    xmid   = (float*)   alloc((size_t)NTOK * DMODEL * 4);
    _Float16* xn16   = (_Float16*)alloc((size_t)NTOK * DMODEL * 2);
    _Float16* h16    = (_Float16*)alloc((size_t)NTOK * DFF * 2);

    auto cvt = [&](const float* src, _Float16* dst, int n) {
        cvt_f32_f16_kernel<<<(n + 255) / 256, 256, 0, stream>>>(src, dst, n);
    };
    auto gemm = [&](const _Float16* A, int lda, const _Float16* W, int ldw,
                    const float* bias, const float* resid,
                    float* Cf, _Float16* Ch, int ldc,
                    int M, int N, int K, int act) {
        dim3 grid((N + 127) / 128, (M + 127) / 128);
        wmma_gemm_kernel<<<grid, 256, 0, stream>>>(A, lda, W, ldw, bias, resid,
                                                   Cf, Ch, ldc, M, N, K, act);
    };

    // weight conversions (fp32 -> fp16)
    cvt(in_proj,  w_in16, 2 * DINNER * DMODEL);
    cvt(x_proj,   w_xp16, XDBL_W * DINNER);
    cvt(dt_proj,  w_dt16, DINNER * DTRANK);
    cvt(out_proj, w_op16, DMODEL * DINNER);
    cvt(ffn_w1,   w_f116, DFF * DMODEL);
    cvt(ffn_w2,   w_f216, DMODEL * DFF);

    // LN1 -> xp16
    ln_kernel<<<NTOK, 256, 0, stream>>>(x, ln1_g, ln1_b, xp16, DMODEL);

    // xz = xp @ in_proj^T  [4096, 3072]
    gemm(xp16, DMODEL, w_in16, DMODEL, nullptr, nullptr,
         xz, nullptr, 2 * DINNER, NTOK, 2 * DINNER, DMODEL, 0);

    // causal conv + SiLU -> u
    {
        int n = NTOK * DINNER;
        conv_silu_kernel<<<(n + 255) / 256, 256, 0, stream>>>(xz, conv_w, conv_b,
                                                              u32, u16);
    }

    // x_dbl = u @ x_proj^T  [4096, 80]  (f32 for scan, f16 for dt_proj)
    gemm(u16, DINNER, w_xp16, DINNER, nullptr, nullptr,
         xdbl, xdbl16, XDBL_W, NTOK, XDBL_W, DINNER, 0);

    // delta = softplus(dt @ dt_proj^T + dt_b)  [4096, 1536]
    gemm(xdbl16, XDBL_W, w_dt16, DTRANK, dt_b, nullptr,
         delta, nullptr, DINNER, NTOK, DINNER, DTRANK, 2);

    // selective scan + skip + SiLU(z) gating -> y16
    {
        int n = BATCH * DINNER;
        scan_kernel<<<(n + 255) / 256, 256, 0, stream>>>(delta, u32, xdbl, xz,
                                                         A_log, Dp, y16);
    }

    // xmid = y @ out_proj^T + x  [4096, 1024]
    gemm(y16, DINNER, w_op16, DINNER, nullptr, x,
         xmid, nullptr, DMODEL, NTOK, DMODEL, DINNER, 0);

    // LN2 -> xn16
    ln_kernel<<<NTOK, 256, 0, stream>>>(xmid, ln2_g, ln2_b, xn16, DMODEL);

    // h = gelu(xn @ w1^T + b1)  [4096, 2048] (f16 only)
    gemm(xn16, DMODEL, w_f116, DMODEL, ffn_b1, nullptr,
         nullptr, h16, DFF, NTOK, DFF, DMODEL, 1);

    // out = h @ w2^T + b2 + xmid  [4096, 1024]
    gemm(h16, DFF, w_f216, DFF, ffn_b2, xmid,
         out, nullptr, DMODEL, NTOK, DMODEL, DFF, 0);
}
